// Pathfinder_72859825209820
// MI455X (gfx1250) — compile-verified, matrix operand order checked by reading, not run on hardware
//
#include <hip/hip_runtime.h>
#include <hip/hip_bf16.h>
#include <stdint.h>

#define DIMC   512
#define LSEQ   2048
#define LPAD   2050
#define NBATCH 32
#define NDEPTH 8
#define KCONV  1536
#define NTOK   (NBATCH * LSEQ)

typedef __attribute__((ext_vector_type(16))) __bf16 v16bf;
typedef __attribute__((ext_vector_type(8)))  float  v8f;
typedef __attribute__((ext_vector_type(8)))  int    v8i;
typedef __attribute__((ext_vector_type(4)))  unsigned int u32x4;
typedef __attribute__((ext_vector_type(8)))  int    i32x8;
typedef __attribute__((ext_vector_type(4)))  int    i32x4;

union AB16 { v16bf v; uint4 q[2]; };
union AB8  { v8i   v; uint2 q[4]; };

__device__ __forceinline__ float bf2f(uint16_t h) {
    return __uint_as_float(((unsigned)h) << 16);
}
__device__ __forceinline__ uint16_t f2bf(float f) {
    unsigned u = __float_as_uint(f);
    unsigned r = u + 0x7FFFu + ((u >> 16) & 1u);
    return (uint16_t)(r >> 16);
}
__device__ __forceinline__ float gelu_exact(float x) {
    return 0.5f * x * (1.0f + erff(x * 0.70710678118654752440f));
}

#if defined(__AMDGCN__) && __has_builtin(__builtin_amdgcn_tensor_load_to_lds)
#define HAVE_TDM 1
__device__ __forceinline__ void tdm_load_2d(unsigned lds_addr, const void* gptr,
                                            int dsz_log,
                                            unsigned tile_d0, unsigned tile_d1,
                                            unsigned tdim0, unsigned tdim1,
                                            unsigned long long stride0_elems) {
    unsigned long long ga = (unsigned long long)(uintptr_t)gptr;
    u32x4 g0;
    g0[0] = 1u;
    g0[1] = lds_addr;
    g0[2] = (unsigned)(ga & 0xFFFFFFFFull);
    g0[3] = (unsigned)((ga >> 32) & 0x01FFFFFFull) | 0x80000000u;
    i32x8 g1;
    g1[0] = (dsz_log & 3) << 16;
    g1[1] = (int)((tdim0 & 0xFFFFu) << 16);
    g1[2] = (int)((tdim0 >> 16) | ((tdim1 & 0xFFFFu) << 16));
    g1[3] = (int)(((tdim1 >> 16) & 0xFFFFu) | ((tile_d0 & 0xFFFFu) << 16));
    g1[4] = (int)(tile_d1 & 0xFFFFu);
    g1[5] = (int)(unsigned)(stride0_elems & 0xFFFFFFFFull);
    g1[6] = (int)(unsigned)((stride0_elems >> 32) & 0xFFFFull);
    g1[7] = 0;
    i32x4 g2 = {0, 0, 0, 0};
    i32x4 g3 = {0, 0, 0, 0};
#if defined(__clang_major__) && __clang_major__ >= 23
    i32x8 g4 = {0, 0, 0, 0, 0, 0, 0, 0};
    __builtin_amdgcn_tensor_load_to_lds(g0, g1, g2, g3, g4, 0);
#else
    __builtin_amdgcn_tensor_load_to_lds(g0, g1, g2, g3, 0);
#endif
}
#endif

__global__ __launch_bounds__(256) void k_embed(const int* __restrict__ xd,
                                               const float* __restrict__ iw,
                                               const float* __restrict__ ib,
                                               uint16_t* __restrict__ hb) {
    size_t i = (size_t)blockIdx.x * 256 + threadIdx.x;
    int c = (int)(i & (DIMC - 1));
    size_t tok = i >> 9;
    int b = (int)(tok >> 11);
    int l = (int)(tok & (LSEQ - 1));
    float x0 = (float)xd[b * LSEQ + l];
    float xm = (l > 0)        ? (float)xd[b * LSEQ + l - 1] : 0.0f;
    float xp = (l < LSEQ - 1) ? (float)xd[b * LSEQ + l + 1] : 0.0f;
    float t0 = x0;
    float t1 = x0 + xp;
    float t2 = x0 + xm;
    float v = iw[c * 3 + 0] * t0 + iw[c * 3 + 1] * t1 + iw[c * 3 + 2] * t2 + ib[c];
    hb[((size_t)(b * LPAD + l + 1)) * DIMC + c] = f2bf(v);
}

__global__ __launch_bounds__(256) void k_repack_conv(const float* __restrict__ cw,
                                                     const float* __restrict__ bg,
                                                     const float* __restrict__ bv,
                                                     uint16_t* __restrict__ wB) {
    size_t i = (size_t)blockIdx.x * 256 + threadIdx.x;
    int kk = (int)(i % KCONV);
    size_t rest = i / KCONV;
    int co = (int)(rest % DIMC);
    int d  = (int)(rest / DIMC);
    int k  = kk >> 9;
    int ci = kk & (DIMC - 1);
    float a = bg[d * DIMC + co] * rsqrtf(bv[d * DIMC + co] + 1e-5f);
    float w = cw[(((size_t)d * DIMC + co) * DIMC + ci) * 3 + k];
    wB[i] = f2bf(w * a);
}

__global__ __launch_bounds__(256) void k_bnc(const float* __restrict__ cb,
                                             const float* __restrict__ bg,
                                             const float* __restrict__ bb,
                                             const float* __restrict__ bm,
                                             const float* __restrict__ bv,
                                             float* __restrict__ bnC) {
    int i = blockIdx.x * 256 + threadIdx.x;
    float a = bg[i] * rsqrtf(bv[i] + 1e-5f);
    bnC[i] = (cb[i] - bm[i]) * a + bb[i];
}

__global__ __launch_bounds__(256) void k_qproj(const float* __restrict__ pw,
                                               int8_t* __restrict__ wq,
                                               float* __restrict__ invw) {
    const int d = blockIdx.x;
    const size_t n = (size_t)DIMC * DIMC;
    const float* w = pw + (size_t)d * n;
    __shared__ float red[256];
    __shared__ float sc;
    float s = 0.0f;
    for (size_t i = threadIdx.x; i < n; i += 256) s += fabsf(w[i]);
    red[threadIdx.x] = s;
    __syncthreads();
    for (int o = 128; o > 0; o >>= 1) {
        if ((int)threadIdx.x < o) red[threadIdx.x] += red[threadIdx.x + o];
        __syncthreads();
    }
    if (threadIdx.x == 0) {
        float cl = fmaxf(red[0] / (float)n, 1e-5f);
        sc = 1.0f / cl;
        invw[d] = cl;
    }
    __syncthreads();
    float scale = sc;
    for (size_t i = threadIdx.x; i < n; i += 256) {
        float q = fminf(fmaxf(rintf(w[i] * scale), -1.0f), 1.0f);
        wq[(size_t)d * n + i] = (int8_t)(int)q;
    }
}

__global__ __launch_bounds__(256) void k_conv_gemm(const uint16_t* __restrict__ hb,
                                                   const uint16_t* __restrict__ wB,
                                                   const float* __restrict__ bnC,
                                                   int8_t* __restrict__ aq,
                                                   float* __restrict__ sxinv) {
    __shared__ uint16_t As[16 * KCONV];
    __shared__ int   smax[16];
    __shared__ float sscale[16];

    const int wave  = threadIdx.x >> 5;
    const int lane  = threadIdx.x & 31;
    const int lrow  = lane & 15;
    const int khalf = (lane >> 4) << 3;
    const int mhi   = (lane >> 4) << 3;
    const int t0    = blockIdx.x * 16;

    const int b0 = t0 >> 11;
    const int l0 = t0 & (LSEQ - 1);
    const uint16_t* gA = hb + (size_t)(b0 * LPAD + l0) * DIMC;

#ifdef HAVE_TDM
    if (threadIdx.x < 32) {
        tdm_load_2d((unsigned)(uintptr_t)As, gA,  1,
                     KCONV, 16,  KCONV, 16,  DIMC);
        __builtin_amdgcn_s_wait_tensorcnt(0);
    }
#else
    for (int i = threadIdx.x; i < 16 * (KCONV / 8); i += 256) {
        int r = i / (KCONV / 8);
        int c = i % (KCONV / 8);
        *(uint4*)(As + r * KCONV + c * 8) = *(const uint4*)(gA + (size_t)r * DIMC + c * 8);
    }
#endif

    const int ncol0 = wave * 64;
    const uint16_t* brow[4];
#pragma unroll
    for (int tt = 0; tt < 4; ++tt) {
        brow[tt] = wB + (size_t)(ncol0 + tt * 16 + lrow) * KCONV;
        __builtin_prefetch(brow[tt], 0, 3);
    }
    __syncthreads();

    const uint16_t* alds = As + lrow * KCONV;
    v8f acc[4] = {};
    for (int k0 = 0; k0 < KCONV; k0 += 32) {
        AB16 a;
        a.q[0] = *(const uint4*)(alds + k0 + khalf);
        a.q[1] = *(const uint4*)(alds + k0 + khalf + 16);
#pragma unroll
        for (int tt = 0; tt < 4; ++tt) {
            AB16 bm;
            bm.q[0] = *(const uint4*)(brow[tt] + k0 + khalf);
            bm.q[1] = *(const uint4*)(brow[tt] + k0 + khalf + 16);
            acc[tt] = __builtin_amdgcn_wmma_f32_16x16x32_bf16(
                false, a.v, false, bm.v, (short)0, acc[tt], false, false);
        }
    }

    if (threadIdx.x < 16) smax[threadIdx.x] = 0;
    __syncthreads();

    float g[4][8];
#pragma unroll
    for (int tt = 0; tt < 4; ++tt) {
        int co = ncol0 + tt * 16 + lrow;
        float c0 = bnC[co];
#pragma unroll
        for (int r = 0; r < 8; ++r) {
            float y = acc[tt][r] + c0;
            float ge = gelu_exact(y);
            g[tt][r] = ge;
            atomicMax(&smax[r + mhi], __float_as_int(fabsf(ge)));
        }
    }
    __syncthreads();
    if (threadIdx.x < 16) {
        float mx = fmaxf(__int_as_float(smax[threadIdx.x]), 1e-5f);
        sscale[threadIdx.x] = 127.0f / mx;
        sxinv[t0 + threadIdx.x] = mx / 127.0f;
    }
    __syncthreads();
#pragma unroll
    for (int tt = 0; tt < 4; ++tt) {
        int co = ncol0 + tt * 16 + lrow;
#pragma unroll
        for (int r = 0; r < 8; ++r) {
            int m = r + mhi;
            float q = fminf(fmaxf(rintf(g[tt][r] * sscale[m]), -128.0f), 127.0f);
            aq[(size_t)(t0 + m) * DIMC + co] = (int8_t)(int)q;
        }
    }
}

__global__ __launch_bounds__(256) void k_bitlin_gemm(const int8_t* __restrict__ aq,
                                                     const float* __restrict__ sxinv,
                                                     const int8_t* __restrict__ wq,
                                                     const float* __restrict__ invw,
                                                     uint16_t* __restrict__ hb) {
    __shared__ int8_t As8[16 * DIMC];

    const int wave  = threadIdx.x >> 5;
    const int lane  = threadIdx.x & 31;
    const int lrow  = lane & 15;
    const int khalf = (lane >> 4) << 3;
    const int mhi   = (lane >> 4) << 3;
    const int t0    = blockIdx.x * 16;

    const int8_t* gA = aq + (size_t)t0 * DIMC;

#ifdef HAVE_TDM
    if (threadIdx.x < 32) {
        tdm_load_2d((unsigned)(uintptr_t)As8, gA,  0,
                     DIMC, 16,  DIMC, 16,  DIMC);
        __builtin_amdgcn_s_wait_tensorcnt(0);
    }
#else
    for (int i = threadIdx.x; i < 16 * (DIMC / 16); i += 256) {
        int r = i / (DIMC / 16);
        int c = i % (DIMC / 16);
        *(uint4*)(As8 + r * DIMC + c * 16) = *(const uint4*)(gA + (size_t)r * DIMC + c * 16);
    }
#endif

    const int ncol0 = wave * 64;
    const int8_t* brow[4];
#pragma unroll
    for (int tt = 0; tt < 4; ++tt) {
        brow[tt] = wq + (size_t)(ncol0 + tt * 16 + lrow) * DIMC;
        __builtin_prefetch(brow[tt], 0, 3);
    }
    __syncthreads();

    const int8_t* alds = As8 + lrow * DIMC;
    v8i acc[4] = {};
    for (int k0 = 0; k0 < DIMC; k0 += 64) {
        AB8 a;
        a.q[0] = *(const uint2*)(alds + k0 + khalf);
        a.q[1] = *(const uint2*)(alds + k0 + khalf + 16);
        a.q[2] = *(const uint2*)(alds + k0 + khalf + 32);
        a.q[3] = *(const uint2*)(alds + k0 + khalf + 48);
#pragma unroll
        for (int tt = 0; tt < 4; ++tt) {
            AB8 bm;
            bm.q[0] = *(const uint2*)(brow[tt] + k0 + khalf);
            bm.q[1] = *(const uint2*)(brow[tt] + k0 + khalf + 16);
            bm.q[2] = *(const uint2*)(brow[tt] + k0 + khalf + 32);
            bm.q[3] = *(const uint2*)(brow[tt] + k0 + khalf + 48);
            acc[tt] = __builtin_amdgcn_wmma_i32_16x16x64_iu8(
                true, a.v, true, bm.v, acc[tt], false, false);
        }
    }

    const float winv = *invw;
#pragma unroll
    for (int tt = 0; tt < 4; ++tt) {
        int co = ncol0 + tt * 16 + lrow;
#pragma unroll
        for (int r = 0; r < 8; ++r) {
            int m = r + mhi;
            int tok = t0 + m;
            float val = (float)acc[tt][r] * sxinv[tok] * winv;
            float ge = gelu_exact(val);
            int bb = tok >> 11;
            int ll = tok & (LSEQ - 1);
            size_t idx = (size_t)(bb * LPAD + ll + 1) * DIMC + co;
            float old = bf2f(hb[idx]);
            hb[idx] = f2bf(old + ge);
        }
    }
}

__global__ __launch_bounds__(256) void k_pool(const uint16_t* __restrict__ hb,
                                              float* __restrict__ pooled) {
    int i = blockIdx.x * 256 + threadIdx.x;
    int c = i & (DIMC - 1);
    int b = i >> 9;
    const uint16_t* base = hb + (size_t)(b * LPAD + 1) * DIMC + c;
    float s = 0.0f;
    for (int l = 0; l < LSEQ; ++l) s += bf2f(base[(size_t)l * DIMC]);
    pooled[i] = s * (1.0f / (float)LSEQ);
}

__global__ __launch_bounds__(256) void k_head(const float* __restrict__ pooled,
                                              const float* __restrict__ hw,
                                              float* __restrict__ out) {
    __shared__ float red[256];
    __shared__ float wqv[512];
    __shared__ float sv[2];
    int tid = threadIdx.x;
    red[tid] = fabsf(hw[tid]) + fabsf(hw[tid + 256]);
    __syncthreads();
    for (int o = 128; o > 0; o >>= 1) { if (tid < o) red[tid] += red[tid + o]; __syncthreads(); }
    if (tid == 0) sv[0] = fmaxf(red[0] * (1.0f / 512.0f), 1e-5f);
    __syncthreads();
    float wsc = 1.0f / sv[0];
    wqv[tid]       = fminf(fmaxf(rintf(hw[tid] * wsc), -1.0f), 1.0f);
    wqv[tid + 256] = fminf(fmaxf(rintf(hw[tid + 256] * wsc), -1.0f), 1.0f);
    __syncthreads();
    for (int b = 0; b < NBATCH; ++b) {
        const float* row = pooled + b * DIMC;
        float a0 = row[tid], a1 = row[tid + 256];
        red[tid] = fmaxf(fabsf(a0), fabsf(a1));
        __syncthreads();
        for (int o = 128; o > 0; o >>= 1) { if (tid < o) red[tid] = fmaxf(red[tid], red[tid + o]); __syncthreads(); }
        if (tid == 0) sv[1] = fmaxf(red[0], 1e-5f);
        __syncthreads();
        float asc = 127.0f / sv[1];
        float q0 = fminf(fmaxf(rintf(a0 * asc), -128.0f), 127.0f);
        float q1 = fminf(fmaxf(rintf(a1 * asc), -128.0f), 127.0f);
        red[tid] = q0 * wqv[tid] + q1 * wqv[tid + 256];
        __syncthreads();
        for (int o = 128; o > 0; o >>= 1) { if (tid < o) red[tid] += red[tid + o]; __syncthreads(); }
        if (tid == 0) {
            float acc = red[0] * (sv[1] / 127.0f) * sv[0];
            out[b] = 1.0f / (1.0f + expf(-acc));
        }
        __syncthreads();
    }
}

extern "C" void kernel_launch(void* const* d_in, const int* in_sizes, int n_in,
                              void* d_out, int out_size, void* d_ws, size_t ws_size,
                              hipStream_t stream) {
    (void)in_sizes; (void)n_in; (void)out_size; (void)ws_size;
    const int*   x_diff   = (const int*)  d_in[0];
    const float* inproj_w = (const float*)d_in[2];
    const float* inproj_b = (const float*)d_in[3];
    const float* conv_w   = (const float*)d_in[4];
    const float* conv_b   = (const float*)d_in[5];
    const float* bn_g     = (const float*)d_in[6];
    const float* bn_b     = (const float*)d_in[7];
    const float* bn_m     = (const float*)d_in[8];
    const float* bn_v     = (const float*)d_in[9];
    const float* proj_w   = (const float*)d_in[10];
    const float* head_w   = (const float*)d_in[11];
    float* out = (float*)d_out;

    size_t off = 0;
    auto carve = [&](size_t bytes) -> void* {
        off = (off + 255) & ~(size_t)255;
        void* p = (char*)d_ws + off;
        off += bytes;
        return p;
    };
    const size_t hbElems = (size_t)NBATCH * LPAD * DIMC;
    uint16_t* wB     = (uint16_t*)carve((size_t)NDEPTH * DIMC * KCONV * 2);
    float*    bnC    = (float*)   carve((size_t)NDEPTH * DIMC * 4);
    int8_t*   projq  = (int8_t*)  carve((size_t)NDEPTH * DIMC * DIMC);
    float*    projiv = (float*)   carve((size_t)NDEPTH * 4);
    uint16_t* hb     = (uint16_t*)carve(hbElems * 2);
    int8_t*   aq     = (int8_t*)  carve((size_t)NTOK * DIMC);
    float*    sxinv  = (float*)   carve((size_t)NTOK * 4);
    float*    pooled = (float*)   carve((size_t)NBATCH * DIMC * 4);

    hipMemsetAsync(hb, 0, hbElems * 2, stream);

    k_repack_conv<<<(NDEPTH * DIMC * KCONV) / 256, 256, 0, stream>>>(conv_w, bn_g, bn_v, wB);
    k_bnc<<<(NDEPTH * DIMC) / 256, 256, 0, stream>>>(conv_b, bn_g, bn_b, bn_m, bn_v, bnC);
    k_qproj<<<NDEPTH, 256, 0, stream>>>(proj_w, projq, projiv);

    k_embed<<<(int)(((size_t)NTOK * DIMC) / 256), 256, 0, stream>>>(x_diff, inproj_w, inproj_b, hb);

    const int mtiles = NTOK / 16;
    for (int d = 0; d < NDEPTH; ++d) {
        k_conv_gemm<<<mtiles, 256, 0, stream>>>(
            hb, wB + (size_t)d * DIMC * KCONV, bnC + (size_t)d * DIMC, aq, sxinv);
        k_bitlin_gemm<<<mtiles, 256, 0, stream>>>(
            aq, sxinv, projq + (size_t)d * DIMC * DIMC, projiv + d, hb);
    }

    k_pool<<<(NBATCH * DIMC) / 256, 256, 0, stream>>>(hb, pooled);
    k_head<<<1, 256, 0, stream>>>(pooled, head_w, out);
}
